// MemoryAccess_28930899705857
// MI455X (gfx1250) — matrix operand order repaired and checked by reading, not compiled, not run on hardware
//
#include <hip/hip_runtime.h>
#include <math.h>

#define NB       16
#define SLOTS    10000
#define SLOT_SZ  256
#define RSLOTS   3
#define WAVES    4

typedef __attribute__((ext_vector_type(16))) _Float16 v16h;
typedef __attribute__((ext_vector_type(8)))  float    v8f;

// Consume one 32x16 f32 K-chunk tile (in LDS) + A rows from global -> one WMMA.
__device__ __forceinline__ v8f wmma_chunk(const float* __restrict__ X, int K, int k0,
                                          int c, int half,
                                          const float (*__restrict__ tb)[20], v8f acc)
{
  // ---- A fragment: lane half 0 -> K k0+0..7 / k0+16..23, half 1 -> +8 ----
  const int kb = k0 + (half << 3);
  const float* xr = X + (size_t)c * K + kb;
  float4 a0 = *(const float4*)(xr + 0);
  float4 a1 = *(const float4*)(xr + 4);
  float4 a2 = *(const float4*)(xr + 16);
  float4 a3 = *(const float4*)(xr + 20);
  v16h a;
  a[0]=(_Float16)a0.x;  a[1]=(_Float16)a0.y;  a[2]=(_Float16)a0.z;  a[3]=(_Float16)a0.w;
  a[4]=(_Float16)a1.x;  a[5]=(_Float16)a1.y;  a[6]=(_Float16)a1.z;  a[7]=(_Float16)a1.w;
  a[8]=(_Float16)a2.x;  a[9]=(_Float16)a2.y;  a[10]=(_Float16)a2.z; a[11]=(_Float16)a2.w;
  a[12]=(_Float16)a3.x; a[13]=(_Float16)a3.y; a[14]=(_Float16)a3.z; a[15]=(_Float16)a3.w;

  // ---- B fragment: lane half 0 -> K 0..7 / 16..23, half 1 -> +8 (matches A) ----
  const int kstart = half << 3;
  v16h b;
  #pragma unroll
  for (int j = 0; j < 16; ++j)
    b[j] = (_Float16)tb[kstart + (j & 7) + ((j & 8) << 1)][c];

  return __builtin_amdgcn_wmma_f32_16x16x32_f16(false, a, false, b,
                                                (short)0, acc, false, false);
}

// ---------------------------------------------------------------------------
// Wave-level WMMA GEMM:  out(16 x N) = epilogue( X(16 x K) @ W(K x N) + bias )
// Each wave owns one 16-wide N tile; K in chunks of 32 via
// v_wmma_f32_16x16x32_f16. B tiles streamed with double-buffered
// global_load_async_to_lds_b128 (ASYNCcnt), one 64B K-row slice per lane.
// EPI: 0=+bias  1=extra.*(acc+bias)  2=tanh -> fused segmented argmax
//      3=sigmoid  4=relu
// ---------------------------------------------------------------------------
template<int EPI>
__global__ __launch_bounds__(32 * WAVES)
void gemm16_wmma(const float* __restrict__ X, const float* __restrict__ W,
                 const float* __restrict__ bias, const float* __restrict__ extra,
                 float* __restrict__ out, unsigned long long* __restrict__ keys,
                 int K, int N)
{
  __shared__ float tileB[WAVES][2][32][20];  // double-buffered 32 K-rows x 16 cols
  const int lane  = threadIdx.x & 31;
  const int wave  = threadIdx.x >> 5;
  const int ntile = blockIdx.x * WAVES + wave;
  const int Ntiles = (N + 15) >> 4;
  if (ntile >= Ntiles) return;               // wave-uniform exit (EXEC all-1 for WMMA)

  const int col0 = ntile << 4;
  const int c    = lane & 15;                // N within tile; also A row for this lane
  const int half = lane >> 4;
  const bool full = (col0 + 16 <= N) && ((N & 3) == 0);

  v8f acc = {};
  const int nchunks = K >> 5;                // K is always a multiple of 32 here

  if (full) {
    // async DMA one 64-byte K-row slice per lane into LDS, double buffered
    #define ISSUE_ASYNC(KC, BUF)                                                  \
      do {                                                                        \
        const float* _s = W + (size_t)(((KC) << 5) + lane) * N + col0;            \
        unsigned long long _ga = (unsigned long long)(uintptr_t)_s;               \
        unsigned _la = (unsigned)(uintptr_t)&tileB[wave][BUF][lane][0];           \
        asm volatile(                                                             \
          "global_load_async_to_lds_b128 %0, %1, off\n\t"                         \
          "global_load_async_to_lds_b128 %0, %1, off offset:16\n\t"               \
          "global_load_async_to_lds_b128 %0, %1, off offset:32\n\t"               \
          "global_load_async_to_lds_b128 %0, %1, off offset:48"                   \
          :: "v"(_la), "v"(_ga) : "memory");                                      \
      } while (0)

    ISSUE_ASYNC(0, 0);
    for (int kc = 0; kc < nchunks; ++kc) {
      const int cur = kc & 1;
      if (kc + 1 < nchunks) {
        ISSUE_ASYNC(kc + 1, cur ^ 1);
        asm volatile("s_wait_asynccnt 0x4" ::: "memory");  // chunk kc landed
      } else {
        asm volatile("s_wait_asynccnt 0x0" ::: "memory");
      }
      acc = wmma_chunk(X, K, kc << 5, c, half, tileB[wave][cur], acc);
    }
    #undef ISSUE_ASYNC
  } else {
    // partial / unaligned tile (only the tiny N=3 GEMM): branch-free clamped loads
    for (int kc = 0; kc < nchunks; ++kc) {
      const int k0 = kc << 5;
      const float* srow = W + (size_t)(k0 + lane) * N;
      #pragma unroll
      for (int j = 0; j < 16; ++j) {
        int col = col0 + j;
        float vv = srow[(col < N) ? col : (N - 1)];    // unconditional load
        tileB[wave][0][lane][j] = (col < N) ? vv : 0.0f;
      }
      asm volatile("s_wait_dscnt 0" ::: "memory");      // cross-lane LDS RAW
      acc = wmma_chunk(X, K, k0, c, half, tileB[wave][0], acc);
      asm volatile("s_wait_dscnt 0" ::: "memory");      // reads before next stores
    }
  }

  // ---- epilogue; C/D layout: VGPR v -> row v (lanes 0-15) / v+8 (lanes 16-31)
  const int n = col0 + c;
  const float bv = (n < N) ? bias[n] : 0.0f;

  if constexpr (EPI == 2) {
    // 10000 % 16 == 0 -> every tile lies inside exactly one read-slot segment
    const int seg  = col0 / SLOTS;
    const int sidx = n - seg * SLOTS;
    #pragma unroll
    for (int v = 0; v < 8; ++v) {
      float z = tanhf(acc[v] + bv);
      unsigned u = __float_as_uint(z);
      u = (u & 0x80000000u) ? ~u : (u | 0x80000000u);        // orderable float
      unsigned long long key =
          ((unsigned long long)u << 32) |
          (unsigned long long)(0xFFFFFFFFu - (unsigned)sidx); // first-index tiebreak
      #pragma unroll
      for (int off = 1; off < 16; off <<= 1) {               // reduce across N half
        unsigned long long ok = __shfl_xor(key, off, 32);
        key = (ok > key) ? ok : key;
      }
      if (c == 0) {
        int row = v + (half << 3);
        atomicMax(&keys[row * RSLOTS + seg], key);
      }
    }
  } else {
    if (n < N) {
      #pragma unroll
      for (int v = 0; v < 8; ++v) {
        int row = v + (half << 3);
        float z = acc[v] + bv;
        if constexpr (EPI == 1) z *= extra[(size_t)row * N + n];
        if constexpr (EPI == 3) z = 1.0f / (1.0f + expf(-z));
        if constexpr (EPI == 4) z = fmaxf(z, 0.0f);
        out[(size_t)row * N + n] = z;
      }
    }
  }
}

// ---------------------------------------------------------------------------
// Row softmax over C channels, one block per row, in place.
// ---------------------------------------------------------------------------
__global__ void softmax_rows(float* __restrict__ x, int C)
{
  __shared__ float red[256];
  const int t = threadIdx.x;
  x += (size_t)blockIdx.x * C;
  float mx = -INFINITY;
  for (int c = t; c < C; c += 256) mx = fmaxf(mx, x[c]);
  red[t] = mx; __syncthreads();
  for (int s = 128; s > 0; s >>= 1) { if (t < s) red[t] = fmaxf(red[t], red[t + s]); __syncthreads(); }
  mx = red[0]; __syncthreads();
  float sum = 0.0f;
  for (int c = t; c < C; c += 256) { float e = expf(x[c] - mx); x[c] = e; sum += e; }
  red[t] = sum; __syncthreads();
  for (int s = 128; s > 0; s >>= 1) { if (t < s) red[t] += red[t + s]; __syncthreads(); }
  const float inv = 1.0f / red[0];
  for (int c = t; c < C; c += 256) x[c] *= inv;
}

__global__ void k_init(unsigned long long* __restrict__ keys, float* __restrict__ m)
{
  const int t = threadIdx.x;
  if (t < NB * RSLOTS) keys[t] = 0ull;
  for (int i = t; i < NB * SLOT_SZ; i += 256) m[i] = 0.0f;
}

// decode packed argmax keys -> read_w / read_idx, and gather memory slots
__global__ void k_decode_gather(const unsigned long long* __restrict__ keys,
                                const float* __restrict__ memory,
                                float* __restrict__ read_w, int* __restrict__ read_idx,
                                float* __restrict__ r_all)
{
  const int blk = blockIdx.x;            // b * RSLOTS + i
  const int b = blk / RSLOTS;
  const int t = threadIdx.x;
  const unsigned long long key = keys[blk];
  const unsigned u  = (unsigned)(key >> 32);
  const unsigned fb = (u & 0x80000000u) ? (u & 0x7FFFFFFFu) : ~u;
  const int idx = (int)(0xFFFFFFFFu - (unsigned)(key & 0xFFFFFFFFu));
  if (t == 0) { read_w[blk] = __uint_as_float(fb); read_idx[blk] = idx; }
  r_all[blk * SLOT_SZ + t] = memory[((size_t)b * SLOTS + idx) * SLOT_SZ + t];
}

__global__ void k_concat(float* __restrict__ xcat, const float* __restrict__ r_all,
                         const float* __restrict__ inputs, int i)
{
  const int b = blockIdx.x, t = threadIdx.x;
  xcat[b * 768 + t]       = r_all[(b * RSLOTS + i) * SLOT_SZ + t];
  xcat[b * 768 + 256 + t] = inputs[b * 512 + t];
  xcat[b * 768 + 512 + t] = inputs[b * 512 + 256 + t];
}

__global__ void k_gate(float* __restrict__ ycat, const float* __restrict__ upd,
                       const float* __restrict__ r_all, const float* __restrict__ upd_w,
                       const float* __restrict__ m, int i)
{
  const int b = blockIdx.x, t = threadIdx.x;
  const float u  = upd_w[b * RSLOTS + i];
  const float rr = u * upd[b * SLOT_SZ + t] +
                   (1.0f - u) * r_all[(b * RSLOTS + i) * SLOT_SZ + t];
  ycat[b * 512 + t]       = rr;
  ycat[b * 512 + 256 + t] = m[b * SLOT_SZ + t];
}

__global__ void k_scale(float* __restrict__ m, const float* __restrict__ t_am,
                        const float* __restrict__ read_w, int i)
{
  const int b = blockIdx.x, t = threadIdx.x;
  m[b * SLOT_SZ + t] = read_w[b * RSLOTS + i] * t_am[b * SLOT_SZ + t];
}

__global__ void k_tanh_out(float* __restrict__ out, const float* __restrict__ m)
{
  const int b = blockIdx.x, t = threadIdx.x;
  out[b * SLOT_SZ + t] = tanhf(m[b * SLOT_SZ + t]);
}

// ---------------------------------------------------------------------------
static void launch_gemm(int epi, const float* X, const float* W, const float* bias,
                        const float* extra, float* out, unsigned long long* keys,
                        int K, int N, hipStream_t s)
{
  const int Ntiles = (N + 15) / 16;
  dim3 grid((Ntiles + WAVES - 1) / WAVES);
  dim3 block(32 * WAVES);
  switch (epi) {
    case 0: gemm16_wmma<0><<<grid, block, 0, s>>>(X, W, bias, extra, out, keys, K, N); break;
    case 1: gemm16_wmma<1><<<grid, block, 0, s>>>(X, W, bias, extra, out, keys, K, N); break;
    case 2: gemm16_wmma<2><<<grid, block, 0, s>>>(X, W, bias, extra, out, keys, K, N); break;
    case 3: gemm16_wmma<3><<<grid, block, 0, s>>>(X, W, bias, extra, out, keys, K, N); break;
    case 4: gemm16_wmma<4><<<grid, block, 0, s>>>(X, W, bias, extra, out, keys, K, N); break;
  }
}

extern "C" void kernel_launch(void* const* d_in, const int* in_sizes, int n_in,
                              void* d_out, int out_size, void* d_ws, size_t ws_size,
                              hipStream_t stream)
{
  (void)in_sizes; (void)n_in; (void)out_size; (void)ws_size;
  const float* inputs   = (const float*)d_in[0];
  const float* memory   = (const float*)d_in[1];
  const float* fa_r_w1  = (const float*)d_in[2];
  const float* fa_r_b1  = (const float*)d_in[3];
  const float* fa_r_w2  = (const float*)d_in[4];
  const float* fa_r_b2  = (const float*)d_in[5];
  const float* W_read   = (const float*)d_in[6];
  const float* b_read   = (const float*)d_in[7];
  const float* fa_u_w1  = (const float*)d_in[8];
  const float* fa_u_b1  = (const float*)d_in[9];
  const float* fa_u_w2  = (const float*)d_in[10];
  const float* fa_u_b2  = (const float*)d_in[11];
  const float* W_uw     = (const float*)d_in[12];
  const float* b_uw     = (const float*)d_in[13];
  const float* fa_um_w1 = (const float*)d_in[14];
  const float* fa_um_b1 = (const float*)d_in[15];
  const float* fa_um_w2 = (const float*)d_in[16];
  const float* fa_um_b2 = (const float*)d_in[17];
  const float* W_um     = (const float*)d_in[18];
  const float* b_um     = (const float*)d_in[19];
  const float* fa_am_w1 = (const float*)d_in[20];
  const float* fa_am_b1 = (const float*)d_in[21];
  const float* fa_am_w2 = (const float*)d_in[22];
  const float* fa_am_b2 = (const float*)d_in[23];
  const float* W_am     = (const float*)d_in[24];
  const float* b_am     = (const float*)d_in[25];
  float* out = (float*)d_out;

  // workspace layout (floats); total ~280 KB -> L2 resident
  float* ws     = (float*)d_ws;
  float* tmpA   = ws + 0;       // 16*768
  float* tmpB   = ws + 12288;   // 16*768
  float* xcat   = ws + 24576;   // 16*768
  float* ycat   = ws + 36864;   // 16*512
  float* r_all  = ws + 45056;   // 16*3*256
  float* mbuf   = ws + 57344;   // 16*256
  float* updb   = ws + 61440;   // 16*256
  float* t_am   = ws + 65536;   // 16*256
  float* read_w = ws + 69632;   // 48
  float* upd_w  = ws + 69680;   // 48
  unsigned long long* keys = (unsigned long long*)(ws + 69728); // 48 u64 (8B aligned)
  int*   read_idx = (int*)(ws + 69824);                         // 48

  k_init<<<1, 256, 0, stream>>>(keys, mbuf);

  // ---- read path: fa_r(inputs) @ W_read -> fused tanh + segmented argmax ----
  launch_gemm(0, inputs, fa_r_w1, fa_r_b1, nullptr, tmpA, nullptr, 512, 512, stream);
  softmax_rows<<<16, 256, 0, stream>>>(tmpA, 512);
  launch_gemm(1, tmpA, fa_r_w2, fa_r_b2, inputs, tmpB, nullptr, 512, 512, stream);
  launch_gemm(2, tmpB, W_read, b_read, nullptr, nullptr, keys, 512, 30000, stream);
  k_decode_gather<<<NB * RSLOTS, SLOT_SZ, 0, stream>>>(keys, memory, read_w, read_idx, r_all);

  // ---- update-weight path: sigmoid(fa_u(inputs) @ W_uw + b_uw) ----
  launch_gemm(0, inputs, fa_u_w1, fa_u_b1, nullptr, tmpA, nullptr, 512, 512, stream);
  softmax_rows<<<16, 256, 0, stream>>>(tmpA, 512);
  launch_gemm(1, tmpA, fa_u_w2, fa_u_b2, inputs, tmpB, nullptr, 512, 512, stream);
  launch_gemm(3, tmpB, W_uw, b_uw, nullptr, upd_w, nullptr, 512, 3, stream);

  // ---- sequential memory-update loop ----
  for (int i = 0; i < RSLOTS; ++i) {
    k_concat<<<16, 256, 0, stream>>>(xcat, r_all, inputs, i);
    launch_gemm(0, xcat, fa_um_w1, fa_um_b1, nullptr, tmpA, nullptr, 768, 768, stream);
    softmax_rows<<<16, 256, 0, stream>>>(tmpA, 768);
    launch_gemm(1, tmpA, fa_um_w2, fa_um_b2, xcat, tmpB, nullptr, 768, 768, stream);
    launch_gemm(4, tmpB, W_um, b_um, nullptr, updb, nullptr, 768, 256, stream);
    k_gate<<<16, 256, 0, stream>>>(ycat, updb, r_all, upd_w, mbuf, i);
    launch_gemm(0, ycat, fa_am_w1, fa_am_b1, nullptr, tmpA, nullptr, 512, 512, stream);
    softmax_rows<<<16, 256, 0, stream>>>(tmpA, 512);
    launch_gemm(1, tmpA, fa_am_w2, fa_am_b2, ycat, tmpB, nullptr, 512, 512, stream);
    launch_gemm(4, tmpB, W_am, b_am, nullptr, t_am, nullptr, 512, 256, stream);
    k_scale<<<16, 256, 0, stream>>>(mbuf, t_am, read_w, i);
  }
  k_tanh_out<<<16, 256, 0, stream>>>(out, mbuf);
}
